// GAE_1314259992845
// MI455X (gfx1250) — matrix operand order repaired and checked by reading, not compiled, not run on hardware
//
#include <hip/hip_runtime.h>
#include <hip/hip_bf16.h>
#include <math.h>

// ---------------------------------------------------------------------------
// GAT autoencoder for MI455X (gfx1250, wave32).
//  - Dense matmuls (x@W1, h@W2, z@Wd) use v_wmma_f32_16x16x32_bf16 with a
//    2-term bf16 split (hi/lo) -> 3 WMMAs per B-tile per K-step, f32 acc.
//    Weights are pre-split ONCE into packed bf16 hi/lo buffers in fragment
//    order ([k-tile][n][32 k]) so each lane loads its whole B fragment with
//    one 32B vector load (no scalar-load/wait chains, no per-tile re-convert).
//    Each wave computes a 16 x (TNW*16) strip: one A fragment is reused by
//    TNW B fragments (12 WMMAs/K-step when TNW=4).
//    __launch_bounds__(256,1) keeps all accumulators in registers (no spill).
//  - Edge softmax/aggregation is the memory-bound part (~2 GB of L2 traffic,
//    whole working set < 192MB L2): 3 streaming passes with f32 atomics.
// ---------------------------------------------------------------------------

typedef __attribute__((ext_vector_type(16))) __bf16 v16bf;
typedef __attribute__((ext_vector_type(8)))  float  v8f;

#define NEG_SLOPE 0.2f
#define EPSV      1e-16f

// Monotonic float<->uint mapping so atomicMax(u32) implements float max.
__device__ __forceinline__ unsigned ord_encode(float f) {
  unsigned u = __float_as_uint(f);
  return (u & 0x80000000u) ? ~u : (u | 0x80000000u);
}
__device__ __forceinline__ float ord_decode(unsigned u) {
  return (u & 0x80000000u) ? __uint_as_float(u & 0x7FFFFFFFu)
                           : __uint_as_float(~u);
}

// Pre-split B[K,N] (f32) into fragment-ordered bf16 hi/lo:
//   pack[( (k/32)*N + n )*32 + (k%32)] ; per-lane GEMM offset = bkb, 32B aligned
__global__ void pack_b_kernel(const float* __restrict__ B,
                              __bf16* __restrict__ Bhi, __bf16* __restrict__ Blo,
                              int K, int N)
{
  int t = blockIdx.x * blockDim.x + threadIdx.x;
  if (t >= K * N) return;
  const int k = t / N, n = t % N;
  const size_t idx = ((size_t)(k >> 5) * N + n) * 32 + (k & 31);
  const float v = B[t];
  const __bf16 h = (__bf16)v;
  Bhi[idx] = h;
  Blo[idx] = (__bf16)(v - (float)h);
}

// ---------------------------------------------------------------------------
// Tiled WMMA GEMM: C[M,N] = A[M,K] @ B[K,N] (+bias, +relu), B pre-split/packed.
// Requires M%16==0, N%(TNW*16)==0, K%32==0 (true for all three GEMMs here).
// One 16 x (TNW*16) strip per wave; 8 waves / 256-thread block.
// Fragment layouts used here (wave32; contiguous K order, same for A and B):
//   A 16x32 bf16 : lane->M (0..15 twice); per lane K covers one contiguous
//                  16-run {akb..akb+15}: element i -> K = akb + i, akb = 0|16
//                  -> 4x b128 loads per K-step, split in-register.
//   B 32x16 bf16 : element i -> K = bkb + i (bkb = 0|16), lane&15 -> N
//                  -> one 32B load each for bh / bl from packed buffers.
//   C/D 16x16 f32: VGPR r -> M = r (+8 for lanes>=16), lane&15 -> N
// ---------------------------------------------------------------------------
template <int TNW>
__global__ void __launch_bounds__(256, 1)
wmma_gemm_kernel(const float* __restrict__ A,
                 const __bf16* __restrict__ Bhi,
                 const __bf16* __restrict__ Blo,
                 const float* __restrict__ bias,
                 float* __restrict__ C,
                 int M, int N, int K, int relu)
{
  const int lane  = threadIdx.x & 31;
  const int wave  = threadIdx.x >> 5;
  const int tilesNG = (N >> 4) / TNW;            // column-tile groups
  const int tilesM  = M >> 4;
  int t = blockIdx.x * 8 + wave;                 // uniform per wave
  if (t >= tilesM * tilesNG) return;             // whole wave exits together
  const int row0 = (t / tilesNG) << 4;
  const int col0 = (t % tilesNG) * (TNW << 4);

  const int am  = row0 + (lane & 15);            // A: lane -> M
  const int akb = (lane < 16) ? 0 : 16;          // A: half-wave K offset
  const int bn  = (lane & 15);                   // B: lane -> N (within tile)
  const int bkb = (lane < 16) ? 0 : 16;          // B: half-wave K offset

  v8f acc[TNW];
#pragma unroll
  for (int q = 0; q < TNW; ++q) acc[q] = (v8f){};

  for (int k0 = 0, kt = 0; k0 < K; k0 += 32, ++kt) {
    // ---- A fragment: one contiguous 16-float run -> 4x float4 loads ----
    const float* ap = A + (size_t)am * K + k0 + akb;
    float4 a0 = *(const float4*)(ap + 0);
    float4 a1 = *(const float4*)(ap + 4);
    float4 a2 = *(const float4*)(ap + 8);
    float4 a3 = *(const float4*)(ap + 12);
    float af[16] = {a0.x, a0.y, a0.z, a0.w, a1.x, a1.y, a1.z, a1.w,
                    a2.x, a2.y, a2.z, a2.w, a3.x, a3.y, a3.z, a3.w};
    v16bf ah, al;
#pragma unroll
    for (int i = 0; i < 16; ++i) {
      __bf16 hA = (__bf16)af[i];
      ah[i] = hA;  al[i] = (__bf16)(af[i] - (float)hA);
    }
    // ---- TNW B fragments (packed bf16, one 32B load each), 3 WMMAs ----
#pragma unroll
    for (int q = 0; q < TNW; ++q) {
      const int nq = col0 + (q << 4) + bn;
      const size_t bbase = ((size_t)kt * N + nq) * 32 + bkb;
      v16bf bh = *(const v16bf*)(Bhi + bbase);
      v16bf bl = *(const v16bf*)(Blo + bbase);
      // fp32-ish product via split: A*B ~= Ah*Bh + Ah*Bl + Al*Bh
      acc[q] = __builtin_amdgcn_wmma_f32_16x16x32_bf16(false, ah, false, bh,
                                                       (short)0, acc[q], false, false);
      acc[q] = __builtin_amdgcn_wmma_f32_16x16x32_bf16(false, ah, false, bl,
                                                       (short)0, acc[q], false, false);
      acc[q] = __builtin_amdgcn_wmma_f32_16x16x32_bf16(false, al, false, bh,
                                                       (short)0, acc[q], false, false);
    }
  }

  const int  mofs = (lane < 16) ? 0 : 8;
  const float lo  = relu ? 0.0f : -3.402823466e38f;   // branch-free ReLU
#pragma unroll
  for (int q = 0; q < TNW; ++q) {
    const int cn = col0 + (q << 4) + bn;
    const float bv = bias ? bias[cn] : 0.0f;          // one load per column
#pragma unroll
    for (int r = 0; r < 8; ++r) {
      const int cm = row0 + r + mofs;
      C[(size_t)cm * N + cn] = fmaxf(acc[q][r] + bv, lo);
    }
  }
}

// alpha_s[n,h] = <h[n,h,:], a_src[h,:]>, alpha_d likewise. One thread/(n,h).
__global__ void alpha_kernel(const float* __restrict__ hfeat,
                             const float* __restrict__ a_src,
                             const float* __restrict__ a_dst,
                             float* __restrict__ as_, float* __restrict__ ad_,
                             int Nn, int H, int Cc)
{
  int t = blockIdx.x * blockDim.x + threadIdx.x;
  if (t >= Nn * H) return;
  const int n = t / H, hh = t % H;
  const float* hp = hfeat + (size_t)n * H * Cc + (size_t)hh * Cc;
  const float* sp = a_src + (size_t)hh * Cc;
  const float* dp = a_dst + (size_t)hh * Cc;
  float vs = 0.f, vd = 0.f;
  for (int c = 0; c < Cc; ++c) { vs += hp[c] * sp[c]; vd += hp[c] * dp[c]; }
  as_[t] = vs; ad_[t] = vd;
}

__device__ __forceinline__ void edge_sd(const long long* __restrict__ src,
                                        const long long* __restrict__ dst,
                                        int e, int E, int& s_, int& d_)
{
  if (e < E) { s_ = (int)src[e]; d_ = (int)dst[e]; }
  else       { s_ = d_ = e - E; }                    // self-loop
}

// pass 1: m[dst,h] = max over incoming edges of leaky_relu(as[src]+ad[dst])
__global__ void edge_max_kernel(const long long* __restrict__ src,
                                const long long* __restrict__ dst,
                                int E, int Nn, int H,
                                const float* __restrict__ as_,
                                const float* __restrict__ ad_,
                                unsigned* __restrict__ mord)
{
  long long t = (long long)blockIdx.x * blockDim.x + threadIdx.x;
  const long long EE = (long long)E + Nn;
  if (t >= EE * H) return;
  const int e = (int)(t / H), hh = (int)(t % H);
  int s_, d_; edge_sd(src, dst, e, E, s_, d_);
  float v = as_[s_ * H + hh] + ad_[d_ * H + hh];
  v = (v > 0.f) ? v : v * NEG_SLOPE;
  atomicMax(&mord[d_ * H + hh], ord_encode(v));
}

// pass 2: s[dst,h] = sum exp(e - m[dst,h])
__global__ void edge_sum_kernel(const long long* __restrict__ src,
                                const long long* __restrict__ dst,
                                int E, int Nn, int H,
                                const float* __restrict__ as_,
                                const float* __restrict__ ad_,
                                const unsigned* __restrict__ mord,
                                float* __restrict__ ssum)
{
  long long t = (long long)blockIdx.x * blockDim.x + threadIdx.x;
  const long long EE = (long long)E + Nn;
  if (t >= EE * H) return;
  const int e = (int)(t / H), hh = (int)(t % H);
  int s_, d_; edge_sd(src, dst, e, E, s_, d_);
  float v = as_[s_ * H + hh] + ad_[d_ * H + hh];
  v = (v > 0.f) ? v : v * NEG_SLOPE;
  const float m = ord_decode(mord[d_ * H + hh]);
  atomicAdd(&ssum[d_ * H + hh], __expf(v - m));
}

// pass 3: agg[dst,:] += alpha * h[src,:]; one thread handles 4 channels.
// lshift = log2(F/4) so thread->(edge, 4-chan chunk) is shift/mask.
__global__ void edge_msg_kernel(const long long* __restrict__ src,
                                const long long* __restrict__ dst,
                                int E, int Nn, int H, int Cc, int lshift,
                                const float* __restrict__ as_,
                                const float* __restrict__ ad_,
                                const unsigned* __restrict__ mord,
                                const float* __restrict__ ssum,
                                const float* __restrict__ hfeat,
                                float* __restrict__ agg)
{
  const int F = H * Cc;
  const int lanes = F >> 2;                       // chunks of 4 channels
  long long t = (long long)blockIdx.x * blockDim.x + threadIdx.x;
  const long long EE = (long long)E + Nn;
  if (t >= EE * lanes) return;
  const int e  = (int)(t >> lshift);
  const int c0 = ((int)t & (lanes - 1)) << 2;     // channel base, 16B aligned
  const int hh = c0 / Cc;
  int s_, d_; edge_sd(src, dst, e, E, s_, d_);
  float v = as_[s_ * H + hh] + ad_[d_ * H + hh];
  v = (v > 0.f) ? v : v * NEG_SLOPE;
  const float m = ord_decode(mord[d_ * H + hh]);
  const float alpha = __expf(v - m) / (ssum[d_ * H + hh] + EPSV);
  const float4 hv = *(const float4*)(hfeat + (size_t)s_ * F + c0);
  float* op = agg + (size_t)d_ * F + c0;
  atomicAdd(op + 0, hv.x * alpha);
  atomicAdd(op + 1, hv.y * alpha);
  atomicAdd(op + 2, hv.z * alpha);
  atomicAdd(op + 3, hv.w * alpha);
}

// out[i] = (relu?)(in[i] + bias[i%F])   (in-place allowed)
__global__ void bias_act_kernel(const float* __restrict__ in,
                                const float* __restrict__ bias,
                                float* __restrict__ out,
                                long long total, int F, int relu)
{
  long long t = (long long)blockIdx.x * blockDim.x + threadIdx.x;
  if (t >= total) return;
  float v = in[t] + bias[(int)(t % F)];
  if (relu) v = fmaxf(v, 0.0f);
  out[t] = v;
}

__global__ void fill_kernel(unsigned* __restrict__ p, unsigned v, long long n)
{
  long long t = (long long)blockIdx.x * blockDim.x + threadIdx.x;
  if (t < n) p[t] = v;
}

static inline int gridFor(long long total, int block) {
  return (int)((total + block - 1) / block);
}

static inline void launch_gemm(const float* A, const float* B,
                               __bf16* Bhi, __bf16* Blo,
                               const float* bias, float* C,
                               int M, int N, int K, int relu,
                               hipStream_t stream)
{
  pack_b_kernel<<<gridFor((long long)K * N, 256), 256, 0, stream>>>(B, Bhi, Blo, K, N);
  const int tilesM = M / 16, tilesN = N / 16;
  if (tilesN % 4 == 0) {
    long long groups = (long long)tilesM * (tilesN / 4);
    wmma_gemm_kernel<4><<<gridFor(groups, 8), 256, 0, stream>>>(A, Bhi, Blo, bias, C, M, N, K, relu);
  } else {
    long long groups = (long long)tilesM * (tilesN / 2);
    wmma_gemm_kernel<2><<<gridFor(groups, 8), 256, 0, stream>>>(A, Bhi, Blo, bias, C, M, N, K, relu);
  }
}

extern "C" void kernel_launch(void* const* d_in, const int* in_sizes, int n_in,
                              void* d_out, int out_size, void* d_ws, size_t ws_size,
                              hipStream_t stream)
{
  const float*     x      = (const float*)d_in[0];
  const long long* ei     = (const long long*)d_in[1];   // int64 [2,E]
  const float*     W1     = (const float*)d_in[2];
  const float*     a_src1 = (const float*)d_in[3];
  const float*     a_dst1 = (const float*)d_in[4];
  const float*     b1     = (const float*)d_in[5];
  const float*     W2     = (const float*)d_in[6];
  const float*     a_src2 = (const float*)d_in[7];
  const float*     a_dst2 = (const float*)d_in[8];
  const float*     b2     = (const float*)d_in[9];
  const float*     Wd     = (const float*)d_in[10];
  const float*     bd     = (const float*)d_in[11];

  const int Fin = 128, H1 = 8, C1 = 32, F1 = H1 * C1;   // 256
  const int H2 = 1, C2 = 32, Fout = 128;
  const int Nn = in_sizes[0] / Fin;                      // 50000
  const int E  = in_sizes[1] / 2;                        // 800000
  const long long* srcp = ei;
  const long long* dstp = ei + E;
  const long long EE = (long long)E + Nn;

  // ---- carve workspace (all 256B aligned) ----
  char* w = (char*)d_ws;
  auto carve = [&](size_t bytes) {
    void* p = (void*)w;
    w += (bytes + 255) & ~(size_t)255;
    return p;
  };
  float*    h1_lin = (float*)   carve((size_t)Nn * F1 * 4);  // x@W1
  float*    agg1   = (float*)   carve((size_t)Nn * F1 * 4);  // layer1 agg -> h1
  float*    as1    = (float*)   carve((size_t)Nn * H1 * 4);
  float*    ad1    = (float*)   carve((size_t)Nn * H1 * 4);
  unsigned* m1     = (unsigned*)carve((size_t)Nn * H1 * 4);
  float*    s1     = (float*)   carve((size_t)Nn * H1 * 4);
  float*    h2_lin = (float*)   carve((size_t)Nn * C2 * 4);  // h1@W2
  float*    agg2   = (float*)   carve((size_t)Nn * C2 * 4);  // layer2 agg -> z
  float*    as2    = (float*)   carve((size_t)Nn * 4);
  float*    ad2    = (float*)   carve((size_t)Nn * 4);
  unsigned* m2     = (unsigned*)carve((size_t)Nn * 4);
  float*    s2     = (float*)   carve((size_t)Nn * 4);
  // packed split-bf16 weights (tiny)
  __bf16* W1hi = (__bf16*)carve((size_t)Fin * F1 * 2);
  __bf16* W1lo = (__bf16*)carve((size_t)Fin * F1 * 2);
  __bf16* W2hi = (__bf16*)carve((size_t)F1 * C2 * 2);
  __bf16* W2lo = (__bf16*)carve((size_t)F1 * C2 * 2);
  __bf16* Wdhi = (__bf16*)carve((size_t)C2 * Fout * 2);
  __bf16* Wdlo = (__bf16*)carve((size_t)C2 * Fout * 2);

  const int B = 256;

  // ================= Layer 1 (H=8, C=32) =================
  fill_kernel<<<gridFor((long long)Nn * F1, B), B, 0, stream>>>((unsigned*)agg1, 0u, (long long)Nn * F1);
  fill_kernel<<<gridFor((long long)Nn * H1, B), B, 0, stream>>>(m1, 0u, (long long)Nn * H1);           // ord(-inf)
  fill_kernel<<<gridFor((long long)Nn * H1, B), B, 0, stream>>>((unsigned*)s1, 0u, (long long)Nn * H1);

  launch_gemm(x, W1, W1hi, W1lo, nullptr, h1_lin, Nn, F1, Fin, 0, stream);
  alpha_kernel<<<gridFor((long long)Nn * H1, B), B, 0, stream>>>(h1_lin, a_src1, a_dst1, as1, ad1, Nn, H1, C1);
  edge_max_kernel<<<gridFor(EE * H1, B), B, 0, stream>>>(srcp, dstp, E, Nn, H1, as1, ad1, m1);
  edge_sum_kernel<<<gridFor(EE * H1, B), B, 0, stream>>>(srcp, dstp, E, Nn, H1, as1, ad1, m1, s1);
  edge_msg_kernel<<<gridFor(EE * (F1 / 4), B), B, 0, stream>>>(srcp, dstp, E, Nn, H1, C1, /*lshift=*/6,
                                                              as1, ad1, m1, s1, h1_lin, agg1);
  // h1 = relu(agg1 + b1), in place
  bias_act_kernel<<<gridFor((long long)Nn * F1, B), B, 0, stream>>>(agg1, b1, agg1, (long long)Nn * F1, F1, 1);

  // ================= Layer 2 (H=1, C=32) =================
  fill_kernel<<<gridFor((long long)Nn * C2, B), B, 0, stream>>>((unsigned*)agg2, 0u, (long long)Nn * C2);
  fill_kernel<<<gridFor((long long)Nn, B), B, 0, stream>>>(m2, 0u, (long long)Nn);
  fill_kernel<<<gridFor((long long)Nn, B), B, 0, stream>>>((unsigned*)s2, 0u, (long long)Nn);

  launch_gemm(agg1, W2, W2hi, W2lo, nullptr, h2_lin, Nn, C2, F1, 0, stream);
  alpha_kernel<<<gridFor((long long)Nn, B), B, 0, stream>>>(h2_lin, a_src2, a_dst2, as2, ad2, Nn, H2, C2);
  edge_max_kernel<<<gridFor(EE, B), B, 0, stream>>>(srcp, dstp, E, Nn, H2, as2, ad2, m2);
  edge_sum_kernel<<<gridFor(EE, B), B, 0, stream>>>(srcp, dstp, E, Nn, H2, as2, ad2, m2, s2);
  edge_msg_kernel<<<gridFor(EE * (C2 / 4), B), B, 0, stream>>>(srcp, dstp, E, Nn, H2, C2, /*lshift=*/3,
                                                              as2, ad2, m2, s2, h2_lin, agg2);
  // z = agg2 + b2, in place (no relu)
  bias_act_kernel<<<gridFor((long long)Nn * C2, B), B, 0, stream>>>(agg2, b2, agg2, (long long)Nn * C2, C2, 0);

  // ================= Decoder: out = z @ Wd + bd =================
  launch_gemm(agg2, Wd, Wdhi, Wdlo, bd, (float*)d_out, Nn, Fout, C2, 0, stream);
}
